// HSVSSBlock_11819749998667
// MI455X (gfx1250) — matrix operand order repaired and checked by reading, not compiled, not run on hardware
//
#include <hip/hip_runtime.h>

typedef __attribute__((ext_vector_type(16))) _Float16 v16h;
typedef __attribute__((ext_vector_type(8)))  float    v8f;
typedef __attribute__((ext_vector_type(4)))  unsigned int u32x4;
typedef __attribute__((ext_vector_type(8)))  int      i32x8;
typedef __attribute__((ext_vector_type(4)))  int      i32x4;

#define Cc   96
#define DIc  192
#define Rr   6
#define Ss   16
#define Kd   4
#define Hh   56
#define Ww   56
#define HWc  3136
#define Lc   49
#define NPc  256
#define NPOS 12544
#define XDc  38

// 2D tile load via Tensor Data Mover: rows x rowlen f32 elements, row stride in elements.
// Issued by one wave (EXEC-independent); caller waits with s_wait_tensorcnt then barriers.
__device__ __forceinline__ void tdm_load_2d(unsigned lds_off, const void* gptr,
                                            unsigned rows, unsigned rowlen,
                                            unsigned rowstride) {
  unsigned long long ga = (unsigned long long)gptr;
  u32x4 g0;
  g0[0] = 1u;                                            // count=1, user mode
  g0[1] = lds_off;                                       // lds_addr (bytes)
  g0[2] = (unsigned)ga;                                  // global_addr[31:0]
  g0[3] = (unsigned)((ga >> 32) & 0x01FFFFFFu) | (2u << 30);  // addr[56:32] | type=2
  i32x8 g1;
  g1[0] = (int)(2u << 16);                               // data_size=2 -> 4 bytes
  g1[1] = (int)((rowlen & 0xFFFFu) << 16);               // tensor_dim0[15:0]
  g1[2] = (int)(((rowlen >> 16) & 0xFFFFu) | ((rows & 0xFFFFu) << 16)); // dim0 hi | dim1 lo
  g1[3] = (int)(((rows >> 16) & 0xFFFFu) | ((rowlen & 0xFFFFu) << 16)); // dim1 hi | tile_dim0
  g1[4] = (int)(rows & 0xFFFFu);                         // tile_dim1 | tile_dim2=0
  g1[5] = (int)rowstride;                                // tensor_dim0_stride[31:0]
  g1[6] = 0;
  g1[7] = 0;
  i32x4 z = {0, 0, 0, 0};
#if __clang_major__ >= 23
  i32x8 z8 = {0, 0, 0, 0, 0, 0, 0, 0};
  __builtin_amdgcn_tensor_load_to_lds(g0, g1, z, z, z8, 0);
#else
  __builtin_amdgcn_tensor_load_to_lds(g0, g1, z, z, 0);
#endif
}

__device__ __forceinline__ void pos2hw(int branch, int p, int l, int& n, int& h, int& w) {
  n = p >> 6;
  int rem = p & 63;
  int q = rem >> 3;
  int r = rem & 7;
  int i = l / 7, j = l % 7;
  if (branch == 0) { h = q * 7 + i; w = r * 7 + j; }
  else             { h = i * 8 + q; w = j * 8 + r; }
}

__device__ __forceinline__ int lmap(int k, int l) {
  int t = (k >= 2) ? (48 - l) : l;
  if (k & 1) return (t % 7) * 7 + (t / 7);
  return t;
}

__device__ __forceinline__ float sigmoidf_(float x) { return 1.f / (1.f + __expf(-x)); }

// ---------------- LayerNorm kernels (channel dim = 96, NCHW) ----------------
__global__ void k_ln0(const float* __restrict__ x, const float* __restrict__ w,
                      const float* __restrict__ b, float* __restrict__ xn,
                      float* __restrict__ acc) {
  int t = blockIdx.x * blockDim.x + threadIdx.x;
  if (t >= NPOS) return;
  int n = t / HWc, pos = t % HWc;
  const float* xp = x + (size_t)n * Cc * HWc + pos;
  float mu = 0.f;
  for (int c = 0; c < Cc; ++c) mu += xp[(size_t)c * HWc];
  mu *= (1.f / Cc);
  float var = 0.f;
  for (int c = 0; c < Cc; ++c) { float d = xp[(size_t)c * HWc] - mu; var += d * d; }
  var *= (1.f / Cc);
  float rstd = rsqrtf(var + 1e-5f);
  float* xno = xn  + (size_t)n * Cc * HWc + pos;
  float* ao  = acc + (size_t)n * Cc * HWc + pos;
  for (int c = 0; c < Cc; ++c) {
    float v = xp[(size_t)c * HWc];
    xno[(size_t)c * HWc] = (v - mu) * rstd * w[c] + b[c];
    ao[(size_t)c * HWc]  = v;
  }
}

__global__ void k_ln_add(const float* __restrict__ yin, const float* __restrict__ w,
                         const float* __restrict__ b, float* __restrict__ acc) {
  int t = blockIdx.x * blockDim.x + threadIdx.x;
  if (t >= NPOS) return;
  int n = t / HWc, pos = t % HWc;
  const float* yp = yin + (size_t)n * Cc * HWc + pos;
  float mu = 0.f;
  for (int c = 0; c < Cc; ++c) mu += yp[(size_t)c * HWc];
  mu *= (1.f / Cc);
  float var = 0.f;
  for (int c = 0; c < Cc; ++c) { float d = yp[(size_t)c * HWc] - mu; var += d * d; }
  var *= (1.f / Cc);
  float rstd = rsqrtf(var + 1e-5f);
  float* ap = acc + (size_t)n * Cc * HWc + pos;
  for (int c = 0; c < Cc; ++c)
    ap[(size_t)c * HWc] += (yp[(size_t)c * HWc] - mu) * rstd * w[c] + b[c];
}

__global__ void k_ln_out(const float* __restrict__ yin, const float* __restrict__ w,
                         const float* __restrict__ b, float* __restrict__ out) {
  int t = blockIdx.x * blockDim.x + threadIdx.x;
  if (t >= NPOS) return;
  int n = t / HWc, pos = t % HWc;
  const float* yp = yin + (size_t)n * Cc * HWc + pos;
  float mu = 0.f;
  for (int c = 0; c < Cc; ++c) mu += yp[(size_t)c * HWc];
  mu *= (1.f / Cc);
  float var = 0.f;
  for (int c = 0; c < Cc; ++c) { float d = yp[(size_t)c * HWc] - mu; var += d * d; }
  var *= (1.f / Cc);
  float rstd = rsqrtf(var + 1e-5f);
  float* op = out + (size_t)n * Cc * HWc + pos;
  for (int c = 0; c < Cc; ++c)
    op[(size_t)c * HWc] = (yp[(size_t)c * HWc] - mu) * rstd * w[c] + b[c];
}

// ---- in_proj WMMA GEMM: (12544 x 96) * (96 x 384); weights TDM-staged in LDS ----
__global__ void __launch_bounds__(128) k_inproj(const float* __restrict__ xn,
                                                const float* __restrict__ in_w,
                                                float* __restrict__ xz, int branch) {
  __shared__ float smemB[64 * Cc];  // 64 output rows (d) x 96 (c), 24 KB
  int ng = blockIdx.x % 6;
  int mt = blockIdx.x / 6;
  int wave = threadIdx.x >> 5;
  int lane = threadIdx.x & 31;
  const float* wsrc = in_w + (size_t)branch * 384 * Cc + (size_t)ng * 64 * Cc;
  if (threadIdx.x < 32) {
    tdm_load_2d((unsigned)(unsigned long long)(void*)smemB, wsrc, 64u, (unsigned)Cc, (unsigned)Cc);
    __builtin_amdgcn_s_wait_tensorcnt(0);
  }
  __syncthreads();

  int nt = ng * 4 + wave;
  int m = lane & 15, hf = lane >> 4;
  int mposA = mt * 16 + m;
  int pA = mposA / Lc, lA = mposA % Lc;
  int n, h, w;
  pos2hw(branch, pA, lA, n, h, w);
  const float* arow = xn + (size_t)n * Cc * HWc + (size_t)h * Ww + w;
  const float* brow = smemB + (size_t)(wave * 16 + m) * Cc;

  v8f acc = {};
  for (int kk = 0; kk < Cc; kk += 32) {
    v16h a, bb;
#pragma unroll
    for (int i = 0; i < 8; ++i) {
      int ka = ((i < 4) ? (2 * i) : (16 + 2 * (i - 4))) + 8 * hf + kk;
      a[2 * i]     = (_Float16)arow[(size_t)ka * HWc];
      a[2 * i + 1] = (_Float16)arow[(size_t)(ka + 1) * HWc];
      int kb = ((i < 4) ? (2 * i) : (16 + 2 * (i - 4))) + 8 * hf + kk;
      bb[2 * i]     = (_Float16)brow[kb];
      bb[2 * i + 1] = (_Float16)brow[kb + 1];
    }
    acc = __builtin_amdgcn_wmma_f32_16x16x32_f16(false, a, false, bb, (short)0, acc, false, false);
  }
  int dcol = nt * 16 + m;
#pragma unroll
  for (int r = 0; r < 8; ++r) {
    int mrow = mt * 16 + r + 8 * hf;
    int p = mrow / Lc, l = mrow % Lc;
    xz[((size_t)p * 384 + dcol) * Lc + l] = acc[r];
  }
}

// ---------------- depthwise 3x3 conv + SiLU on 7x7 grid ----------------
__global__ void k_dwconv(const float* __restrict__ xz, const float* __restrict__ conv_w,
                         const float* __restrict__ conv_b, float* __restrict__ xc, int branch) {
  int t = blockIdx.x * blockDim.x + threadIdx.x;
  if (t >= NPc * DIc * Lc) return;
  int l = t % Lc;
  int rest = t / Lc;
  int d = rest % DIc;
  int p = rest / DIc;
  int i = l / 7, j = l % 7;
  const float* in = xz + ((size_t)p * 384 + d) * Lc;
  const float* wg = conv_w + ((size_t)branch * DIc + d) * 9;
  float s = conv_b[branch * DIc + d];
#pragma unroll
  for (int di = 0; di < 3; ++di) {
    int ii = i + di - 1;
    if (ii < 0 || ii > 6) continue;
#pragma unroll
    for (int dj = 0; dj < 3; ++dj) {
      int jj = j + dj - 1;
      if (jj < 0 || jj > 6) continue;
      s += wg[di * 3 + dj] * in[ii * 7 + jj];
    }
  }
  s = s * sigmoidf_(s);
  xc[((size_t)p * DIc + d) * Lc + l] = s;
}

// ---- x_dbl WMMA GEMM per direction: (12544 x 192) * (192 x 38); weights TDM-staged ----
__global__ void __launch_bounds__(96) k_xdbl(const float* __restrict__ xc,
                                             const float* __restrict__ xproj_w,
                                             float* __restrict__ xdbl, int branch) {
  __shared__ float smemB[XDc * DIc];  // 38 rows (c) x 192 (d), ~29 KB
  int k  = blockIdx.x & 3;
  int mt = blockIdx.x >> 2;
  int wave = threadIdx.x >> 5;      // 0..2 -> nt
  int lane = threadIdx.x & 31;
  const float* wsrc = xproj_w + (((size_t)branch * Kd + k) * XDc) * DIc;
  if (threadIdx.x < 32) {
    tdm_load_2d((unsigned)(unsigned long long)(void*)smemB, wsrc, (unsigned)XDc,
                (unsigned)DIc, (unsigned)DIc);
    __builtin_amdgcn_s_wait_tensorcnt(0);
  }
  __syncthreads();

  int nt = wave;
  int m = lane & 15, hf = lane >> 4;
  int mposA = mt * 16 + m;
  int pA = mposA / Lc, lA = mposA % Lc;
  int lm = lmap(k, lA);
  const float* arow = xc + ((size_t)pA * DIc) * Lc + lm;
  int ccol = nt * 16 + m;
  int crow = (ccol < XDc) ? ccol : 0;   // clamp; out-of-range columns discarded at store
  const float* brow = smemB + (size_t)crow * DIc;

  v8f acc = {};
  for (int kk = 0; kk < DIc; kk += 32) {
    v16h a, bb;
#pragma unroll
    for (int i = 0; i < 8; ++i) {
      int ka = ((i < 4) ? (2 * i) : (16 + 2 * (i - 4))) + 8 * hf + kk;
      a[2 * i]     = (_Float16)arow[(size_t)ka * Lc];
      a[2 * i + 1] = (_Float16)arow[(size_t)(ka + 1) * Lc];
      int kb = ((i < 4) ? (2 * i) : (16 + 2 * (i - 4))) + 8 * hf + kk;
      bb[2 * i]     = (_Float16)brow[kb];
      bb[2 * i + 1] = (_Float16)brow[kb + 1];
    }
    acc = __builtin_amdgcn_wmma_f32_16x16x32_f16(false, a, false, bb, (short)0, acc, false, false);
  }
  if (ccol < XDc) {
#pragma unroll
    for (int r = 0; r < 8; ++r) {
      int mrow = mt * 16 + r + 8 * hf;
      int p = mrow / Lc, l = mrow % Lc;
      xdbl[(((size_t)p * Kd + k) * XDc + ccol) * Lc + l] = acc[r];
    }
  }
}

// ---------------- fused dt-proj + softplus + selective scan ----------------
__global__ void k_scan(const float* __restrict__ xc, const float* __restrict__ xdbl,
                       const float* __restrict__ dt_w, const float* __restrict__ dt_b,
                       const float* __restrict__ A_log, const float* __restrict__ Ds,
                       float* __restrict__ ys, int branch) {
  int t = blockIdx.x * blockDim.x + threadIdx.x;
  if (t >= NPc * Kd * DIc) return;
  int d = t % DIc;
  int rest = t / DIc;
  int k = rest & 3;
  int p = rest >> 2;
  int kd = (branch * Kd + k) * DIc + d;
  float Av[Ss];
#pragma unroll
  for (int s = 0; s < Ss; ++s) Av[s] = -__expf(A_log[(size_t)kd * Ss + s]);
  float wr[Rr];
#pragma unroll
  for (int r = 0; r < Rr; ++r) wr[r] = dt_w[(size_t)kd * Rr + r];
  float bdt = dt_b[kd];
  float Dv = Ds[kd];
  const float* xrow  = xc + ((size_t)p * DIc + d) * Lc;
  const float* dbase = xdbl + (((size_t)p * Kd + k) * XDc) * Lc;
  float* yrow = ys + (((size_t)p * Kd + k) * DIc + d) * Lc;
  float hst[Ss];
#pragma unroll
  for (int s = 0; s < Ss; ++s) hst[s] = 0.f;
  for (int l = 0; l < Lc; ++l) {
    float xv = xrow[lmap(k, l)];
    float dtr = bdt;
#pragma unroll
    for (int r = 0; r < Rr; ++r) dtr += wr[r] * dbase[(size_t)r * Lc + l];
    float dt = (dtr > 20.f) ? dtr : log1pf(__expf(dtr));
    float y = 0.f;
#pragma unroll
    for (int s = 0; s < Ss; ++s) {
      float Bv = dbase[(size_t)(Rr + s) * Lc + l];
      float Cv = dbase[(size_t)(Rr + Ss + s) * Lc + l];
      float dA = __expf(dt * Av[s]);
      hst[s] = hst[s] * dA + dt * xv * Bv;
      y += hst[s] * Cv;
    }
    yrow[l] = y + Dv * xv;
  }
}

// ---------------- cross_merge + LN(DI) + SiLU(z) gate ----------------
__global__ void k_merge(const float* __restrict__ ys, const float* __restrict__ xz,
                        const float* __restrict__ onw, const float* __restrict__ onb,
                        float* __restrict__ yg, int branch) {
  int t = blockIdx.x * blockDim.x + threadIdx.x;
  if (t >= NPOS) return;
  int p = t / Lc, l = t % Lc;
  int idx1 = (l % 7) * 7 + (l / 7);
  const float* base = ys + (size_t)p * Kd * DIc * Lc;
  float* yrow = yg + (size_t)p * DIc * Lc;
  float mu = 0.f;
  for (int d = 0; d < DIc; ++d) {
    float v = base[((size_t)0 * DIc + d) * Lc + l]
            + base[((size_t)1 * DIc + d) * Lc + idx1]
            + base[((size_t)2 * DIc + d) * Lc + (48 - l)]
            + base[((size_t)3 * DIc + d) * Lc + (48 - idx1)];
    yrow[(size_t)d * Lc + l] = v;
    mu += v;
  }
  mu *= (1.f / DIc);
  float var = 0.f;
  for (int d = 0; d < DIc; ++d) { float dv = yrow[(size_t)d * Lc + l] - mu; var += dv * dv; }
  var *= (1.f / DIc);
  float rstd = rsqrtf(var + 1e-5f);
  for (int d = 0; d < DIc; ++d) {
    float v = yrow[(size_t)d * Lc + l];
    float z = xz[((size_t)p * 384 + DIc + d) * Lc + l];
    float g = z * sigmoidf_(z);
    yrow[(size_t)d * Lc + l] = ((v - mu) * rstd * onw[branch * DIc + d] + onb[branch * DIc + d]) * g;
  }
}

// ---- out_proj WMMA GEMM: (12544 x 192) * (192 x 96); weights TDM-staged ----
__global__ void __launch_bounds__(64) k_outproj(const float* __restrict__ yg,
                                                const float* __restrict__ out_w,
                                                float* __restrict__ yb, int branch) {
  __shared__ float smemB[32 * DIc];  // 32 rows (c) x 192 (d), 24 KB
  int ng = blockIdx.x % 3;
  int mt = blockIdx.x / 3;
  int wave = threadIdx.x >> 5;      // 0..1
  int lane = threadIdx.x & 31;
  const float* wsrc = out_w + (size_t)branch * Cc * DIc + (size_t)ng * 32 * DIc;
  if (threadIdx.x < 32) {
    tdm_load_2d((unsigned)(unsigned long long)(void*)smemB, wsrc, 32u, (unsigned)DIc, (unsigned)DIc);
    __builtin_amdgcn_s_wait_tensorcnt(0);
  }
  __syncthreads();

  int nt = ng * 2 + wave;
  int m = lane & 15, hf = lane >> 4;
  int mposA = mt * 16 + m;
  int pA = mposA / Lc, lA = mposA % Lc;
  const float* arow = yg + ((size_t)pA * DIc) * Lc + lA;
  int ccol = nt * 16 + m;
  const float* brow = smemB + (size_t)(wave * 16 + m) * DIc;
  v8f acc = {};
  for (int kk = 0; kk < DIc; kk += 32) {
    v16h a, bb;
#pragma unroll
    for (int i = 0; i < 8; ++i) {
      int ka = ((i < 4) ? (2 * i) : (16 + 2 * (i - 4))) + 8 * hf + kk;
      a[2 * i]     = (_Float16)arow[(size_t)ka * Lc];
      a[2 * i + 1] = (_Float16)arow[(size_t)(ka + 1) * Lc];
      int kb = ((i < 4) ? (2 * i) : (16 + 2 * (i - 4))) + 8 * hf + kk;
      bb[2 * i]     = (_Float16)brow[kb];
      bb[2 * i + 1] = (_Float16)brow[kb + 1];
    }
    acc = __builtin_amdgcn_wmma_f32_16x16x32_f16(false, a, false, bb, (short)0, acc, false, false);
  }
#pragma unroll
  for (int r = 0; r < 8; ++r) {
    int mrow = mt * 16 + r + 8 * hf;
    int p = mrow / Lc, l = mrow % Lc;
    int n, h, w;
    pos2hw(branch, p, l, n, h, w);
    yb[((size_t)n * Cc + ccol) * HWc + (size_t)h * Ww + w] = acc[r];
  }
}

// ---- fc1 WMMA GEMM + bias + tanh-GELU; weights TDM-staged in LDS ----
__global__ void __launch_bounds__(128) k_fc1(const float* __restrict__ h2ln,
                                             const float* __restrict__ fc1_w,
                                             const float* __restrict__ fc1_b,
                                             float* __restrict__ h2) {
  __shared__ float smemB[64 * Cc];
  int ng = blockIdx.x % 6;
  int mt = blockIdx.x / 6;
  int wave = threadIdx.x >> 5;
  int lane = threadIdx.x & 31;
  const float* wsrc = fc1_w + (size_t)ng * 64 * Cc;
  if (threadIdx.x < 32) {
    tdm_load_2d((unsigned)(unsigned long long)(void*)smemB, wsrc, 64u, (unsigned)Cc, (unsigned)Cc);
    __builtin_amdgcn_s_wait_tensorcnt(0);
  }
  __syncthreads();

  int nt = ng * 4 + wave;
  int m = lane & 15, hf = lane >> 4;
  int mposA = mt * 16 + m;
  int n = mposA / HWc, pos = mposA % HWc;
  const float* arow = h2ln + (size_t)n * Cc * HWc + pos;
  const float* brow = smemB + (size_t)(wave * 16 + m) * Cc;
  v8f acc = {};
  for (int kk = 0; kk < Cc; kk += 32) {
    v16h a, bb;
#pragma unroll
    for (int i = 0; i < 8; ++i) {
      int ka = ((i < 4) ? (2 * i) : (16 + 2 * (i - 4))) + 8 * hf + kk;
      a[2 * i]     = (_Float16)arow[(size_t)ka * HWc];
      a[2 * i + 1] = (_Float16)arow[(size_t)(ka + 1) * HWc];
      int kb = ((i < 4) ? (2 * i) : (16 + 2 * (i - 4))) + 8 * hf + kk;
      bb[2 * i]     = (_Float16)brow[kb];
      bb[2 * i + 1] = (_Float16)brow[kb + 1];
    }
    acc = __builtin_amdgcn_wmma_f32_16x16x32_f16(false, a, false, bb, (short)0, acc, false, false);
  }
  int dcol = nt * 16 + m;
#pragma unroll
  for (int r = 0; r < 8; ++r) {
    int mrow = mt * 16 + r + 8 * hf;
    float v = acc[r] + fc1_b[dcol];
    float u = 0.7978845608028654f * (v + 0.044715f * v * v * v);
    h2[(size_t)mrow * 384 + dcol] = 0.5f * v * (1.f + tanhf(u));
  }
}

// ---- fc2 WMMA GEMM + bias + residual -> out; weights TDM-staged ----
__global__ void __launch_bounds__(64) k_fc2(const float* __restrict__ h2,
                                            const float* __restrict__ fc2_w,
                                            const float* __restrict__ fc2_b,
                                            const float* __restrict__ accb,
                                            float* __restrict__ out) {
  __shared__ float smemB[32 * 384];  // 32 rows (c) x 384 (d), 48 KB
  int ng = blockIdx.x % 3;
  int mt = blockIdx.x / 3;
  int wave = threadIdx.x >> 5;      // 0..1
  int lane = threadIdx.x & 31;
  const float* wsrc = fc2_w + (size_t)ng * 32 * 384;
  if (threadIdx.x < 32) {
    tdm_load_2d((unsigned)(unsigned long long)(void*)smemB, wsrc, 32u, 384u, 384u);
    __builtin_amdgcn_s_wait_tensorcnt(0);
  }
  __syncthreads();

  int nt = ng * 2 + wave;
  int m = lane & 15, hf = lane >> 4;
  int mposA = mt * 16 + m;
  const float* arow = h2 + (size_t)mposA * 384;
  int ccol = nt * 16 + m;
  const float* brow = smemB + (size_t)(wave * 16 + m) * 384;
  v8f cacc = {};
  for (int kk = 0; kk < 384; kk += 32) {
    v16h a, bb;
#pragma unroll
    for (int i = 0; i < 8; ++i) {
      int ka = ((i < 4) ? (2 * i) : (16 + 2 * (i - 4))) + 8 * hf + kk;
      a[2 * i]     = (_Float16)arow[ka];
      a[2 * i + 1] = (_Float16)arow[ka + 1];
      int kb = ((i < 4) ? (2 * i) : (16 + 2 * (i - 4))) + 8 * hf + kk;
      bb[2 * i]     = (_Float16)brow[kb];
      bb[2 * i + 1] = (_Float16)brow[kb + 1];
    }
    cacc = __builtin_amdgcn_wmma_f32_16x16x32_f16(false, a, false, bb, (short)0, cacc, false, false);
  }
#pragma unroll
  for (int r = 0; r < 8; ++r) {
    int mrow = mt * 16 + r + 8 * hf;
    int n = mrow / HWc, pos = mrow % HWc;
    size_t idx = ((size_t)n * Cc + ccol) * HWc + pos;
    out[idx] = accb[idx] + cacc[r] + fc2_b[ccol];
  }
}

extern "C" void kernel_launch(void* const* d_in, const int* in_sizes, int n_in,
                              void* d_out, int out_size, void* d_ws, size_t ws_size,
                              hipStream_t stream) {
  (void)in_sizes; (void)n_in; (void)out_size; (void)ws_size;
  const float* x       = (const float*)d_in[0];
  const float* ln0_w   = (const float*)d_in[1];
  const float* ln0_b   = (const float*)d_in[2];
  const float* lnl_w   = (const float*)d_in[3];
  const float* lnl_b   = (const float*)d_in[4];
  const float* lng_w   = (const float*)d_in[5];
  const float* lng_b   = (const float*)d_in[6];
  const float* ln2_w   = (const float*)d_in[7];
  const float* ln2_b   = (const float*)d_in[8];
  const float* in_w    = (const float*)d_in[9];
  const float* conv_w  = (const float*)d_in[10];
  const float* conv_b  = (const float*)d_in[11];
  const float* xproj_w = (const float*)d_in[12];
  const float* dt_w    = (const float*)d_in[13];
  const float* dt_b    = (const float*)d_in[14];
  const float* A_log   = (const float*)d_in[15];
  const float* Ds      = (const float*)d_in[16];
  const float* onorm_w = (const float*)d_in[17];
  const float* onorm_b = (const float*)d_in[18];
  const float* out_w   = (const float*)d_in[19];
  const float* fc1_w   = (const float*)d_in[20];
  const float* fc1_b   = (const float*)d_in[21];
  const float* fc2_w   = (const float*)d_in[22];
  const float* fc2_b   = (const float*)d_in[23];

  float* ws   = (float*)d_ws;
  float* xn   = ws;                   // 1204224
  float* acc  = xn   + 1204224;       // 1204224
  float* xz   = acc  + 1204224;       // 4816896
  float* xc   = xz   + 4816896;       // 2408448
  float* xdbl = xc   + 2408448;       // 1906688
  float* ys   = xdbl + 1906688;       // 9633792
  float* yg   = ys   + 9633792;       // 2408448
  float* yb   = yg   + 2408448;       // 1204224
  float* h2ln = xc;                   // reuse (branch buffers free by MLP time)
  float* h2   = xz;                   // reuse

  k_ln0<<<(NPOS + 255) / 256, 256, 0, stream>>>(x, ln0_w, ln0_b, xn, acc);

  for (int br = 0; br < 2; ++br) {
    k_inproj <<<784 * 6, 128, 0, stream>>>(xn, in_w, xz, br);
    k_dwconv <<<(NPc * DIc * Lc + 255) / 256, 256, 0, stream>>>(xz, conv_w, conv_b, xc, br);
    k_xdbl   <<<784 * 4, 96, 0, stream>>>(xc, xproj_w, xdbl, br);
    k_scan   <<<(NPc * Kd * DIc + 255) / 256, 256, 0, stream>>>(xc, xdbl, dt_w, dt_b, A_log, Ds, ys, br);
    k_merge  <<<(NPOS + 127) / 128, 128, 0, stream>>>(ys, xz, onorm_w, onorm_b, yg, br);
    k_outproj<<<784 * 3, 64, 0, stream>>>(yg, out_w, yb, br);
    k_ln_add <<<(NPOS + 255) / 256, 256, 0, stream>>>(yb, br ? lng_w : lnl_w, br ? lng_b : lnl_b, acc);
  }

  k_ln_out<<<(NPOS + 255) / 256, 256, 0, stream>>>(acc, ln2_w, ln2_b, h2ln);
  k_fc1   <<<784 * 6, 128, 0, stream>>>(h2ln, fc1_w, fc1_b, h2);
  k_fc2   <<<784 * 3, 64, 0, stream>>>(h2, fc2_w, fc2_b, acc, (float*)d_out);
}
